// DiffeqSolver_1984274890990
// MI455X (gfx1250) — compile-verified, matrix operand order checked by reading, not run on hardware
//
#include <hip/hip_runtime.h>
#include <cstddef>

typedef _Float16 v16h __attribute__((ext_vector_type(16)));
typedef _Float16 v8h  __attribute__((ext_vector_type(8)));
typedef float    v8f  __attribute__((ext_vector_type(8)));

constexpr int S_ = 3, B_ = 1024, D_ = 32, H_ = 128, T_ = 256;
constexpr int ROWS = S_ * B_;       // 3072 independent ODE states
constexpr int NBLOCKS = ROWS / 16;  // 192 blocks; 1 block = 2 waves = one 16-row tile

// LDS-only split barrier: the workgroup barriers here protect only the LDS
// staging tiles, so wait on DScnt alone (NOT storecnt/loadcnt -- keeps the
// per-step HBM output stores off the serial critical path).
__device__ __forceinline__ void lds_barrier() {
    asm volatile(
        "s_wait_dscnt 0x0\n\t"
        "s_barrier_signal -1\n\t"
        "s_barrier_wait -1"
        ::: "memory");
}

// Branch-free saturating tanh: 1 - 2/(e^{2x}+1), e^{2x} = 2^(2*log2e*x).
// v_mul + v_exp_f32 + v_add + v_rcp_f32 + v_fma (vs ~10-op IEEE div expansion).
__device__ __forceinline__ float fast_tanh(float x) {
    float e = __builtin_amdgcn_exp2f(x * 2.885390081777927f);
    return 1.0f - 2.0f * __builtin_amdgcn_rcpf(e + 1.0f);
}

__device__ __forceinline__ v8f wmma_f16(v16h a, v16h b, v8f c) {
    // (neg_a, A, neg_b, B, c_mod, C, reuse_a, reuse_b)
    return __builtin_amdgcn_wmma_f32_16x16x32_f16(false, a, false, b,
                                                  (short)0, c, false, false);
}

// Load one 16x32 f16 A/B fragment from a row-major [row][K] LDS tile.
// lane 0-15 : M/N = lane,    K = 0..7  then 16..23  (kb = 0)
// lane 16-31: M/N = lane-16, K = 8..15 then 24..31  (kb = 8)   (ISA 7.12.2)
__device__ __forceinline__ v16h load_frag(const _Float16* p, int row,
                                          int rstride, int kb) {
    const _Float16* q = p + row * rstride + kb;
    v8h lo = *(const v8h*)(q);
    v8h hi = *(const v8h*)(q + 16);
    return __builtin_shufflevector(lo, hi, 0, 1, 2, 3, 4, 5, 6, 7,
                                           8, 9, 10, 11, 12, 13, 14, 15);
}

extern "C" __global__ __launch_bounds__(64)
void rk4_node_kernel(const float* __restrict__ fp, const float* __restrict__ ts,
                     const float* __restrict__ W1, const float* __restrict__ b1,
                     const float* __restrict__ W2, const float* __restrict__ b2,
                     float* __restrict__ out) {
    // Transposed f16 weights: w1t[n][k] (128x32), w2t[n][k] (32x128)
    __shared__ __align__(16) _Float16 w1t[H_ * D_];
    __shared__ __align__(16) _Float16 w2t[D_ * H_];
    __shared__ __align__(16) _Float16 sy[16 * D_];   // y staging  [m][k=32]
    __shared__ __align__(16) _Float16 sh[16 * H_];   // h staging  [m][k=128]

    const int tid    = threadIdx.x;       // 0..63
    const int wave   = tid >> 5;          // 0/1: owns N-half of H and of D
    const int lane   = tid & 31;
    const int lane16 = lane & 15;
    const int rowhi  = lane >> 4;         // C-layout: component r -> m = r + 8*rowhi
    const int kb     = rowhi * 8;         // A/B-layout K base for this lane half

    // ---- stage weights (fp32 global -> f16 transposed LDS), both waves ----
    for (int i = tid; i < D_ * H_; i += 64) {      // W1: [k=32][n=128]
        int k = i >> 7, n = i & 127;
        w1t[n * D_ + k] = (_Float16)W1[i];
    }
    for (int i = tid; i < H_ * D_; i += 64) {      // W2: [k=128][n=32]
        int k = i >> 5, n = i & 31;
        w2t[n * H_ + k] = (_Float16)W2[i];
    }
    lds_barrier();

    // ---- hoist this wave's B fragments into registers (live for all 255 steps)
    v16h w1f[4];                           // GEMM1 N-tiles j = wave*4 + jj
#pragma unroll
    for (int jj = 0; jj < 4; ++jj)
        w1f[jj] = load_frag(w1t, (wave * 4 + jj) * 16 + lane16, D_, kb);
    v16h w2f[4];                           // GEMM2 K-chunks, N-tile nt = wave
#pragma unroll
    for (int kt = 0; kt < 4; ++kt)
        w2f[kt] = load_frag(w2t + kt * 32, wave * 16 + lane16, H_, kb);

    float b1v[4];
#pragma unroll
    for (int jj = 0; jj < 4; ++jj) b1v[jj] = b1[(wave * 4 + jj) * 16 + lane16];
    const float b2v = b2[wave * 16 + lane16];

    const int row0 = blockIdx.x * 16;
    const int ncol = wave * 16 + lane16;   // d column owned by this lane

    // ---- load initial y half in C-layout: comp r -> (m = r+8*rowhi, d = ncol)
    v8f y;
#pragma unroll
    for (int r = 0; r < 8; ++r)
        y[r] = fp[(row0 + r + 8 * rowhi) * D_ + ncol];

    // t = 0 output is the initial state (write-once -> non-temporal)
#pragma unroll
    for (int r = 0; r < 8; ++r)
        __builtin_nontemporal_store(
            y[r], &out[(size_t)((row0 + r + 8 * rowhi) * T_ + 0) * D_ + ncol]);

    // ---- f(y) = tanh(y W1 + b1) W2 + b2, this wave's column-half of result
    auto evalf = [&](v8f yin) -> v8f {
        // stage our half of y as f16, row-major [m][32]
#pragma unroll
        for (int r = 0; r < 8; ++r)
            sy[(r + 8 * rowhi) * D_ + ncol] = (_Float16)yin[r];
        lds_barrier();                                    // bar1 (LDS only)
        v16h ay = load_frag(sy, lane16, D_, kb);          // full 16x32 A frag
        // GEMM1: our 4 of 8 N-tiles; bias via accumulator init; tanh + stage
#pragma unroll
        for (int jj = 0; jj < 4; ++jj) {
            v8f c;
#pragma unroll
            for (int i = 0; i < 8; ++i) c[i] = b1v[jj];
            c = wmma_f16(ay, w1f[jj], c);
            const int nh = (wave * 4 + jj) * 16 + lane16;
#pragma unroll
            for (int r = 0; r < 8; ++r)
                sh[(r + 8 * rowhi) * H_ + nh] = (_Float16)fast_tanh(c[r]);
        }
        lds_barrier();                                    // bar2 (LDS only)
        // GEMM2: chain 4 WMMAs over K=128, our 16-column half of D
        v8f acc;
#pragma unroll
        for (int i = 0; i < 8; ++i) acc[i] = b2v;
#pragma unroll
        for (int kt = 0; kt < 4; ++kt) {
            v16h ha = load_frag(sh + kt * 32, lane16, H_, kb);
            acc = wmma_f16(ha, w2f[kt], acc);
        }
        return acc;
    };

    // ---- sequential RK4 scan over time (the only serial dimension)
    for (int t = 0; t < T_ - 1; ++t) {
        const float dt = ts[t + 1] - ts[t];
        v8f k1 = evalf(y);
        v8f kacc = k1;
        v8f yt;
#pragma unroll
        for (int i = 0; i < 8; ++i) yt[i] = y[i] + 0.5f * dt * k1[i];
        v8f k2 = evalf(yt);
#pragma unroll
        for (int i = 0; i < 8; ++i) {
            kacc[i] += 2.0f * k2[i];
            yt[i] = y[i] + 0.5f * dt * k2[i];
        }
        v8f k3 = evalf(yt);
#pragma unroll
        for (int i = 0; i < 8; ++i) {
            kacc[i] += 2.0f * k3[i];
            yt[i] = y[i] + dt * k3[i];
        }
        v8f k4 = evalf(yt);
        const float s = dt * (1.0f / 6.0f);
#pragma unroll
        for (int i = 0; i < 8; ++i) y[i] += s * (kacc[i] + k4[i]);

#pragma unroll
        for (int r = 0; r < 8; ++r)
            __builtin_nontemporal_store(
                y[r],
                &out[(size_t)((row0 + r + 8 * rowhi) * T_ + (t + 1)) * D_ + ncol]);
    }
}

extern "C" void kernel_launch(void* const* d_in, const int* in_sizes, int n_in,
                              void* d_out, int out_size, void* d_ws, size_t ws_size,
                              hipStream_t stream) {
    const float* fp = (const float*)d_in[0];   // first_point [S,B,D]
    const float* ts = (const float*)d_in[1];   // time_steps [T]
    const float* W1 = (const float*)d_in[2];   // [D,H]
    const float* b1 = (const float*)d_in[3];   // [H]
    const float* W2 = (const float*)d_in[4];   // [H,D]
    const float* b2 = (const float*)d_in[5];   // [D]
    float* out = (float*)d_out;                // [S,B,T,D]

    hipLaunchKernelGGL(rk4_node_kernel, dim3(NBLOCKS), dim3(64), 0, stream,
                       fp, ts, W1, b1, W2, b2, out);
}
